// BRNNCTC_60413009985618
// MI455X (gfx1250) — matrix operand order repaired and checked by reading, not compiled, not run on hardware
//
#include <hip/hip_runtime.h>
#include <math.h>
#include <stdint.h>

typedef __attribute__((ext_vector_type(16))) _Float16 v16h;
typedef __attribute__((ext_vector_type(8)))  float    v8f;
typedef __attribute__((ext_vector_type(4)))  unsigned int v4u;
typedef __attribute__((ext_vector_type(8)))  int      v8i;
typedef __attribute__((ext_vector_type(4)))  int      v4i;

#define B_    32
#define T_    1024
#define F_    128
#define H_    128
#define G4    512      // 4*H
#define ENC_  256
#define INNER 512
#define V_    64
#define L_    200
#define S_    401
#define M_    (B_*T_)
#define NEGF  (-1e30f)

#if defined(__has_builtin)
#if __has_builtin(__builtin_amdgcn_tensor_load_to_lds) && __has_builtin(__builtin_amdgcn_s_wait_tensorcnt)
#define HAVE_TDM 1
#endif
#endif
#ifndef HAVE_TDM
#define HAVE_TDM 0
#endif

__device__ __forceinline__ float sigf(float x) { return 1.0f / (1.0f + __expf(-x)); }

__device__ __forceinline__ float lae(float a, float b) {
    float mx = fmaxf(a, b), mn = fminf(a, b);
    if (mx <= -1e29f) return NEGF;
    return mx + log1pf(__expf(mn - mx));
}

#if HAVE_TDM
// Issue a TDM 2D tile load: 32 rows x 512 f32, row stride T_*G4 elements,
// global tile start gptr -> LDS at lptr. One call per wave issues one DMA.
__device__ __forceinline__ void tdm_load_xw_tile(const float* gptr, float* lptr)
{
    const unsigned long long ga = (unsigned long long)(uintptr_t)gptr;
    const unsigned int lds_addr = (unsigned int)(uintptr_t)lptr; // addr[31:0] = LDS offset

    v4u g0;
    g0[0] = 1u;                                    // count=1 (valid), is_restore=0, gather=0
    g0[1] = lds_addr;                              // lds_addr[31:0]
    g0[2] = (unsigned int)(ga & 0xFFFFFFFFu);      // global_addr[31:0]
    g0[3] = (unsigned int)((ga >> 32) & 0x01FFFFFFu) | (2u << 30); // global_addr[56:32], type=2

    v8i g1;
    g1[0] = (int)(2u << 16);                       // workgroup_mask=0, data_size=2 (4 bytes)
    g1[1] = (int)(512u << 16);                     // tensor_dim0[15:0] in [31:16] = 512
    g1[2] = (int)(32u << 16);                      // tensor_dim0[31:16]=0 | tensor_dim1[15:0]=32
    g1[3] = (int)(512u << 16);                     // tensor_dim1[31:16]=0 | tile_dim0=512
    g1[4] = 32;                                    // tile_dim1=32, tile_dim2=0
    g1[5] = (int)((unsigned)T_ * (unsigned)G4);    // tensor_dim0_stride[31:0] = 524288
    g1[6] = 0;                                     // stride[47:32]=0, dim1_stride lo=0
    g1[7] = 0;

    const v4i gz4 = {0, 0, 0, 0};                  // 2D tensor: groups 2/3 disabled
    const v8i gz8 = {0, 0, 0, 0, 0, 0, 0, 0};
    // 6-arg form (clang-23 / therock-10.0 headers): extra int32x8 group + cpol
    __builtin_amdgcn_tensor_load_to_lds(g0, g1, gz4, gz4, gz8, 0);
}
#endif

// ---------------------------------------------------------------------------
// Generic wave-per-tile WMMA GEMM: out[M,N] = act(A[M,K] @ W[N,K]^T + bias[N])
// A/B converted f32->f16 in registers per the CDNA5 16-bit A/B VGPR striping.
// act: 0 = identity, 1 = tanh
// ---------------------------------------------------------------------------
__global__ void k_gemm(const float* __restrict__ A, const float* __restrict__ W,
                       const float* __restrict__ bias, float* __restrict__ out,
                       int Mdim, int Ndim, int Kdim, int act)
{
    const int lane = threadIdx.x & 31;
    const int wave = threadIdx.x >> 5;
    const int tile = blockIdx.x * 8 + wave;
    const int ntiles = Ndim >> 4;
    const int mt = tile / ntiles;
    const int nt = tile - mt * ntiles;
    const int l15 = lane & 15;
    const bool hi = lane >= 16;

    const size_t arow = (size_t)(mt * 16 + l15) * Kdim;
    const size_t brow = (size_t)(nt * 16 + l15) * Kdim;

    v8f acc;
    const float bv = bias[nt * 16 + l15];
#pragma unroll
    for (int r = 0; r < 8; ++r) acc[r] = bv;

    for (int k0 = 0; k0 < Kdim; k0 += 32) {
        // A fragment: 16x32 f16, lanes 0-15: M=lane, K = k0 + {0..7, 16..23};
        // lanes 16-31: K = k0+8 + {0..7, 16..23}
        const float* ap = A + arow + k0 + (hi ? 8 : 0);
        const float4 a0 = *(const float4*)(ap);
        const float4 a1 = *(const float4*)(ap + 4);
        const float4 a2 = *(const float4*)(ap + 16);
        const float4 a3 = *(const float4*)(ap + 20);
        v16h af;
        af[0]  = (_Float16)a0.x; af[1]  = (_Float16)a0.y; af[2]  = (_Float16)a0.z; af[3]  = (_Float16)a0.w;
        af[4]  = (_Float16)a1.x; af[5]  = (_Float16)a1.y; af[6]  = (_Float16)a1.z; af[7]  = (_Float16)a1.w;
        af[8]  = (_Float16)a2.x; af[9]  = (_Float16)a2.y; af[10] = (_Float16)a2.z; af[11] = (_Float16)a2.w;
        af[12] = (_Float16)a3.x; af[13] = (_Float16)a3.y; af[14] = (_Float16)a3.z; af[15] = (_Float16)a3.w;

        // B fragment: 32x16 f16, lanes 0-15: N=lane, K = k0 + {0..7, 16..23}; lanes 16-31: K = k0+8 + {0..7, 16..23}
        const float* bp = W + brow + k0 + (hi ? 8 : 0);
        const float4 b0 = *(const float4*)(bp);
        const float4 b1 = *(const float4*)(bp + 4);
        const float4 b2 = *(const float4*)(bp + 16);
        const float4 b3 = *(const float4*)(bp + 20);
        v16h bf;
        bf[0]  = (_Float16)b0.x; bf[1]  = (_Float16)b0.y; bf[2]  = (_Float16)b0.z; bf[3]  = (_Float16)b0.w;
        bf[4]  = (_Float16)b1.x; bf[5]  = (_Float16)b1.y; bf[6]  = (_Float16)b1.z; bf[7]  = (_Float16)b1.w;
        bf[8]  = (_Float16)b2.x; bf[9]  = (_Float16)b2.y; bf[10] = (_Float16)b2.z; bf[11] = (_Float16)b2.w;
        bf[12] = (_Float16)b3.x; bf[13] = (_Float16)b3.y; bf[14] = (_Float16)b3.z; bf[15] = (_Float16)b3.w;

        acc = __builtin_amdgcn_wmma_f32_16x16x32_f16(false, af, false, bf,
                                                     (short)0, acc, false, false);
        __builtin_prefetch(ap + 32, 0, 0);   // global_prefetch_b8, next K block
    }

#pragma unroll
    for (int r = 0; r < 8; ++r) {
        float v = acc[r];
        if (act) v = tanhf(v);
        const int mg = mt * 16 + r + (hi ? 8 : 0);
        out[(size_t)mg * Ndim + nt * 16 + l15] = v;
    }
}

// ---------------------------------------------------------------------------
// BiLSTM recurrence. One workgroup (32 waves) per direction; sequential over T.
// Per step: gates[32,512] = xw[:,t,:] + h @ Whh^T via 32 waves x (2 M-tiles x
// 4 K-steps) of v_wmma_f32_16x16x32_f16; h kept in LDS as f16 (A-layout read),
// cell state in LDS f32. The xw[:,t,:] tile (32 rows strided by T*4H) is
// prefetched one step ahead via the Tensor Data Mover into a double-buffered
// LDS tile, synchronized with s_wait_tensorcnt. dir==1 scans t descending.
// ---------------------------------------------------------------------------
__global__ __launch_bounds__(1024) void k_lstm(
    const float* __restrict__ xw_f, const float* __restrict__ xw_b,
    const float* __restrict__ Whh_f, const float* __restrict__ Whh_b,
    float* __restrict__ enc)
{
    extern __shared__ char smem[];
    float*    gates = (float*)smem;                 // 32*512 f32 (64 KB)
    float*    cst   = gates + 32 * G4;              // 32*128 f32 (16 KB)
    _Float16* h16   = (_Float16*)(cst + 32 * H_);   // 32*128 f16 ( 8 KB)
    float*    xb0   = (float*)(h16 + 32 * H_);      // 32*512 f32 (64 KB)
    float*    xb1   = xb0 + 32 * G4;                // 32*512 f32 (64 KB)

    const int dir = blockIdx.x;
    const float* xw  = dir ? xw_b  : xw_f;
    const float* Whh = dir ? Whh_b : Whh_f;

    const int tid  = threadIdx.x;
    const int lane = tid & 31;
    const int wave = tid >> 5;
    const int l15  = lane & 15;
    const bool hi  = lane >= 16;
    const int ncol = wave * 16 + l15;               // wave owns one 16-col N tile

    for (int e = tid; e < 32 * H_; e += 1024) { cst[e] = 0.0f; h16[e] = (_Float16)0.0f; }

#if HAVE_TDM
    // prologue: DMA the first timestep's gate tile into buffer 0
    if (tid < 1) {
        const int t0 = dir ? (T_ - 1) : 0;
        tdm_load_xw_tile(xw + (size_t)t0 * G4, xb0);
        __builtin_amdgcn_s_wait_tensorcnt(0);
    }
#endif
    __syncthreads();

    for (int step = 0; step < T_; ++step) {
        const int t = dir ? (T_ - 1 - step) : step;
        float* xcur = (step & 1) ? xb1 : xb0;

#if HAVE_TDM
        // prefetch next timestep's tile into the other buffer (overlaps WMMAs)
        if (step + 1 < T_ && tid < 1) {
            const int tn = dir ? (T_ - 2 - step) : (step + 1);
            tdm_load_xw_tile(xw + (size_t)tn * G4, (step & 1) ? xb0 : xb1);
        }
#endif

        // init accumulators with precomputed input gates (includes bias)
        v8f c0, c1;
#pragma unroll
        for (int r = 0; r < 8; ++r) {
            const int b0 = r + (hi ? 8 : 0);
#if HAVE_TDM
            c0[r] = xcur[b0 * G4 + ncol];
            c1[r] = xcur[(b0 + 16) * G4 + ncol];
#else
            c0[r] = xw[((size_t)b0 * T_ + t) * G4 + ncol];
            c1[r] = xw[((size_t)(b0 + 16) * T_ + t) * G4 + ncol];
#endif
        }

#pragma unroll
        for (int k0 = 0; k0 < H_; k0 += 32) {
            // B fragment from recurrent weights (global, L2-resident)
            const float* bp = Whh + (size_t)ncol * H_ + k0 + (hi ? 8 : 0);
            const float4 w0 = *(const float4*)(bp);
            const float4 w1 = *(const float4*)(bp + 4);
            const float4 w2 = *(const float4*)(bp + 16);
            const float4 w3 = *(const float4*)(bp + 20);
            v16h bf;
            bf[0]  = (_Float16)w0.x; bf[1]  = (_Float16)w0.y; bf[2]  = (_Float16)w0.z; bf[3]  = (_Float16)w0.w;
            bf[4]  = (_Float16)w1.x; bf[5]  = (_Float16)w1.y; bf[6]  = (_Float16)w1.z; bf[7]  = (_Float16)w1.w;
            bf[8]  = (_Float16)w2.x; bf[9]  = (_Float16)w2.y; bf[10] = (_Float16)w2.z; bf[11] = (_Float16)w2.w;
            bf[12] = (_Float16)w3.x; bf[13] = (_Float16)w3.y; bf[14] = (_Float16)w3.z; bf[15] = (_Float16)w3.w;

            // A fragments (h) from LDS, striped per the 16-bit A layout
            const _Float16* a0p = h16 + l15 * H_ + k0 + (hi ? 8 : 0);
            const _Float16* a1p = h16 + (16 + l15) * H_ + k0 + (hi ? 8 : 0);
            v16h af0, af1;
#pragma unroll
            for (int e = 0; e < 8; ++e) {
                af0[e] = a0p[e];  af0[8 + e] = a0p[16 + e];
                af1[e] = a1p[e];  af1[8 + e] = a1p[16 + e];
            }
            c0 = __builtin_amdgcn_wmma_f32_16x16x32_f16(false, af0, false, bf,
                                                        (short)0, c0, false, false);
            c1 = __builtin_amdgcn_wmma_f32_16x16x32_f16(false, af1, false, bf,
                                                        (short)0, c1, false, false);
        }

#pragma unroll
        for (int r = 0; r < 8; ++r) {
            const int b0 = r + (hi ? 8 : 0);
            gates[b0 * G4 + ncol]        = c0[r];
            gates[(b0 + 16) * G4 + ncol] = c1[r];
        }
        __syncthreads();

        // cell / hidden update: i,f,g,o PyTorch gate order
        for (int e = tid; e < 32 * H_; e += 1024) {
            const int m = e >> 7, j = e & 127;
            const float gi = sigf(gates[m * G4 + j]);
            const float gf = sigf(gates[m * G4 + H_ + j]);
            const float gg = tanhf(gates[m * G4 + 2 * H_ + j]);
            const float go = sigf(gates[m * G4 + 3 * H_ + j]);
            const float c  = gf * cst[e] + gi * gg;
            const float h  = go * tanhf(c);
            cst[e] = c;
            h16[e] = (_Float16)h;
            enc[((size_t)m * T_ + t) * ENC_ + dir * H_ + j] = h;
        }

#if HAVE_TDM
        // ensure next tile has landed before the publish barrier
        if (tid < 1) __builtin_amdgcn_s_wait_tensorcnt(0);
#endif
        __syncthreads();
    }
}

// ---------------------------------------------------------------------------
// Row-wise log_softmax over V=64; one wave per row, wave32 shuffle reductions.
// ---------------------------------------------------------------------------
__global__ void k_lsm(const float* __restrict__ in, float* __restrict__ out)
{
    const int lane = threadIdx.x & 31;
    const int wave = threadIdx.x >> 5;
    const size_t row = (size_t)blockIdx.x * 8 + wave;
    const float x0 = in[row * V_ + lane];
    const float x1 = in[row * V_ + lane + 32];
    float mx = fmaxf(x0, x1);
    for (int o = 16; o > 0; o >>= 1) mx = fmaxf(mx, __shfl_xor(mx, o, 32));
    float sm = __expf(x0 - mx) + __expf(x1 - mx);
    for (int o = 16; o > 0; o >>= 1) sm += __shfl_xor(sm, o, 32);
    const float lse = mx + __logf(sm);
    out[row * V_ + lane]      = x0 - lse;
    out[row * V_ + lane + 32] = x1 - lse;
}

// ---------------------------------------------------------------------------
// CTC forward DP (log space). One block per batch element; alpha double-
// buffered in LDS; 1023 sequential steps with workgroup barriers.
// ---------------------------------------------------------------------------
__global__ void k_ctc(const float* __restrict__ logp, const int* __restrict__ tgt,
                      const int* __restrict__ tlenp, const int* __restrict__ ilenp,
                      float* __restrict__ out_slot)
{
    __shared__ float alpha[2][S_ + 1];
    __shared__ int ext[S_ + 1];
    __shared__ unsigned char skip[S_ + 1];

    const int b = blockIdx.x;
    const int s = threadIdx.x;
    const int ilen = ilenp[b];

    if (s < S_) ext[s] = (s & 1) ? tgt[b * L_ + (s >> 1)] : 0;
    __syncthreads();
    if (s < S_) {
        skip[s] = ((s & 1) && s >= 2 && ext[s] != ext[s - 2]) ? 1 : 0;
        float a = NEGF;
        if (s <= 1) a = logp[((size_t)b * T_) * V_ + ext[s]];
        alpha[0][s] = a;
    }
    __syncthreads();

    int cur = 0;
    for (int t = 1; t < T_; ++t) {
        const int nxt = cur ^ 1;
        if (s < S_) {
            const float a  = alpha[cur][s];
            const float a1 = (s >= 1) ? alpha[cur][s - 1] : NEGF;
            float m = lae(a, a1);
            if (skip[s]) m = lae(m, alpha[cur][s - 2]);
            const float nv = m + logp[((size_t)b * T_ + t) * V_ + ext[s]];
            alpha[nxt][s] = (t < ilen) ? nv : a;
        }
        __syncthreads();
        cur = nxt;
    }

    if (s == 0) {
        const int tl = tlenp[b];
        const int last = 2 * tl;
        const float ll = lae(alpha[cur][last], alpha[cur][last - 1]);
        atomicAdd(out_slot, (-ll / (float)tl) * (1.0f / (float)B_));
    }
}

__global__ void k_zero(float* p, int n)
{
    const int i = threadIdx.x + blockIdx.x * blockDim.x;
    if (i < n) p[i] = 0.0f;
}

// ---------------------------------------------------------------------------
extern "C" void kernel_launch(void* const* d_in, const int* in_sizes, int n_in,
                              void* d_out, int out_size, void* d_ws, size_t ws_size,
                              hipStream_t stream)
{
    const float* x       = (const float*)d_in[0];
    const float* Wih_f   = (const float*)d_in[1];
    const float* Whh_f   = (const float*)d_in[2];
    const float* b_f     = (const float*)d_in[3];
    const float* Wih_b   = (const float*)d_in[4];
    const float* Whh_b   = (const float*)d_in[5];
    const float* b_b     = (const float*)d_in[6];
    const float* W_fwd   = (const float*)d_in[7];
    const float* b_fwd   = (const float*)d_in[8];
    const float* W_base  = (const float*)d_in[9];
    const float* b_base  = (const float*)d_in[10];
    const float* W_rle   = (const float*)d_in[11];
    const float* b_rle   = (const float*)d_in[12];
    const int*   ilen    = (const int*)d_in[13];
    const int*   targets = (const int*)d_in[14];
    const int*   tlen    = (const int*)d_in[15];
    const int*   rles    = (const int*)d_in[16];
    const int*   rlen    = (const int*)d_in[17];
    float* out = (float*)d_out;

    // workspace layout (floats), with reuse after the LSTM stage:
    float* ws   = (float*)d_ws;
    float* xw_f = ws;                                   // [M, 512]
    float* xw_b = ws + (size_t)M_ * G4;                 // [M, 512]
    float* enc  = ws + 2 * (size_t)M_ * G4;             // [M, 256]
    float* fo   = xw_f;                                 // reuse: [M, 512]
    float* lg_b = xw_b;                                 // reuse: [M, 64]
    float* lg_r = xw_b + (size_t)M_ * V_;               // [M, 64]
    float* lp_b = xw_b + 2 * (size_t)M_ * V_;           // [M, 64]
    float* lp_r = xw_b + 3 * (size_t)M_ * V_;           // [M, 64]
    (void)in_sizes; (void)n_in; (void)out_size; (void)ws_size;

    // 1) input-gate GEMMs for both directions (over un-flipped x)
    const int blocks_xw = (M_ / 16) * (G4 / 16) / 8;    // 8192
    k_gemm<<<blocks_xw, 256, 0, stream>>>(x, Wih_f, b_f, xw_f, M_, G4, F_, 0);
    k_gemm<<<blocks_xw, 256, 0, stream>>>(x, Wih_b, b_b, xw_b, M_, G4, F_, 0);

    // 2) bidirectional recurrence (2 workgroups, one per direction)
    const size_t lds_bytes = 32 * G4 * sizeof(float)        // gates
                           + 32 * H_ * sizeof(float)        // cell
                           + 32 * H_ * sizeof(_Float16)     // h (f16)
                           + 2 * 32 * G4 * sizeof(float);   // TDM double buffer
    k_lstm<<<2, 1024, lds_bytes, stream>>>(xw_f, xw_b, Whh_f, Whh_b, enc);

    // 3) projection with tanh, then both heads
    const int blocks_fo = (M_ / 16) * (INNER / 16) / 8;     // 8192
    k_gemm<<<blocks_fo, 256, 0, stream>>>(enc, W_fwd, b_fwd, fo, M_, INNER, ENC_, 1);
    const int blocks_hd = (M_ / 16) * (V_ / 16) / 8;        // 1024
    k_gemm<<<blocks_hd, 256, 0, stream>>>(fo, W_base, b_base, lg_b, M_, V_, INNER, 0);
    k_gemm<<<blocks_hd, 256, 0, stream>>>(fo, W_rle, b_rle, lg_r, M_, V_, INNER, 0);

    // 4) log-softmax over V
    k_lsm<<<M_ / 8, 256, 0, stream>>>(lg_b, lp_b);
    k_lsm<<<M_ / 8, 256, 0, stream>>>(lg_r, lp_r);

    // 5) CTC losses -> d_out[0], d_out[1]
    k_zero<<<1, 32, 0, stream>>>(out, 2);
    k_ctc<<<B_, 512, 0, stream>>>(lp_b, targets, tlen, ilen, out + 0);
    k_ctc<<<B_, 512, 0, stream>>>(lp_r, rles, rlen, ilen, out + 1);
}
